// SO2Convolution_70609262346613
// MI455X (gfx1250) — matrix operand order repaired and checked by reading, not compiled, not run on hardware
//
#include <hip/hip_runtime.h>
#include <hip/hip_bf16.h>
#include <math.h>

// ---------------------------------------------------------------------------
// SO2Convolution fused kernel for MI455X (gfx1250, wave32, WMMA)
//
//   h  = silu(silu(x_edge@W1+b1)@W2+b2)                (fp32 VALU, tiny)
//   out[b,m,d]   = sum_{k<=128,c} h~[b,k] x[b,m,c]  W3~[k,(m,c,d)]
//   gate[b,d]    = sum_{m0<5,k<=128,c} h~[b,k] x[b,m0,c] W3~[k,(m0,c,d)+MAIN]
//   (h~ has a virtual 1.0 row k=128; W3~ row 128 = b3  -> bias folded in)
//
// GEMM form: A[b,r] = h[b,k]*x[b,m,c] built in-register (outer product),
// B[r,d] = W3 repacked to bf16 in exact v_wmma_f32_16x16x32_bf16 fragment
// order. fp32 accumulate. ~4.5M WMMA instrs total. B fragments are
// ping-pong double-buffered so loads for chunk-half i+1 overlap the 4 WMMAs
// of chunk-half i (avoids full s_wait_loadcnt 0x0 drains).
// ---------------------------------------------------------------------------

typedef __attribute__((ext_vector_type(16))) __bf16 v16bf;
typedef __attribute__((ext_vector_type(8)))  float  v8f;

#define E_     4096
#define M_     12
#define C_     64
#define D_     64
#define M0_    5
#define XD_    64
#define EC_    128
#define MAIN_  (M_*C_*D_)            // 49152
#define TOTAL_ (MAIN_ + M0_*C_*XD_) // 69632

#define NCH_M   258              // (129*64)/32 K-chunks per (m, d-block), main
#define NCH_E   1290             // (5*129*64)/32 K-chunks per d-block, extra
#define CHELEMS 512              // 32 lanes * 16 bf16 slots per chunk

// workspace layout (bytes)
#define SZ_H    ((size_t)E_*EC_*4)                 // fp32 h          (2.00 MB)
#define MPELEMS ((size_t)M_*4*NCH_M*CHELEMS)       // 6,340,608 bf16 (12.68 MB)
#define EPELEMS ((size_t)4*NCH_E*CHELEMS)          // 2,641,920 bf16  (5.28 MB)
#define OFF_H   ((size_t)0)
#define OFF_MP  SZ_H
#define OFF_EP  (OFF_MP + MPELEMS*2)

// ---------------------------------------------------------------------------
// Pack W3 (+ b3 as virtual row k=128) into bf16, in the exact per-fragment
// order the WMMA B operand consumes:  [... chunk][lane][slot]
//   B order used here (16-bit, 32x16): lanes 0-15 -> N=lane, K=slot;
//                             lanes 16-31 -> N=lane-16, K=16+slot.
// ---------------------------------------------------------------------------
__global__ void so2_pack_w3(const float* __restrict__ W3,
                            const float* __restrict__ b3,
                            __bf16* __restrict__ MP,
                            __bf16* __restrict__ EP)
{
    size_t p = (size_t)blockIdx.x * blockDim.x + threadIdx.x;
    if (p >= MPELEMS + EPELEMS) return;

    if (p < MPELEMS) {
        int slot  = (int)(p & 15);
        int lane  = (int)((p >> 4) & 31);
        int chunk = (int)((p >> 9) % NCH_M);
        int md    = (int)((p >> 9) / NCH_M);
        int db    = md & 3;
        int m     = md >> 2;
        int Kin   = slot + 16 * (lane >> 4);
        int d     = db * 16 + (lane & 15);
        int r     = chunk * 32 + Kin;      // r = k*64 + c, k in [0,129)
        int k     = r >> 6;
        int c     = r & 63;
        int t     = m * C_ * D_ + c * D_ + d;
        float v   = (k < EC_) ? W3[(size_t)k * TOTAL_ + t] : b3[t];
        MP[p] = (__bf16)v;
    } else {
        size_t q  = p - MPELEMS;
        int slot  = (int)(q & 15);
        int lane  = (int)((q >> 4) & 31);
        int chunk = (int)((q >> 9) % NCH_E);
        int db    = (int)((q >> 9) / NCH_E);
        int Kin   = slot + 16 * (lane >> 4);
        int d     = db * 16 + (lane & 15);
        int r     = chunk * 32 + Kin;      // r = (m0*129 + k)*64 + c
        int m0    = r / (129 * 64);
        int k     = (r >> 6) % 129;
        int c     = r & 63;
        int t     = MAIN_ + m0 * C_ * XD_ + c * XD_ + d;
        float v   = (k < EC_) ? W3[(size_t)k * TOTAL_ + t] : b3[t];
        EP[q] = (__bf16)v;
    }
}

// ---------------------------------------------------------------------------
// Edge MLP: 2 x [4096x128]@[128x128] + SiLU, fp32. 0.27 GFLOP -> negligible;
// keep full precision since h feeds the 73-GFLOP contraction.
// Block = 128 threads handles 32 edges; thread t owns output column j = t.
// ---------------------------------------------------------------------------
__global__ __launch_bounds__(128)
void so2_mlp(const float* __restrict__ XE,
             const float* __restrict__ W1, const float* __restrict__ B1,
             const float* __restrict__ W2, const float* __restrict__ B2,
             float* __restrict__ H)
{
    __shared__ float xe[32][128];
    __shared__ float h1[32][128];
    const int tid = threadIdx.x;
    const int b0  = blockIdx.x * 32;

    for (int i = tid; i < 32 * 128; i += 128)
        xe[i >> 7][i & 127] = XE[(size_t)b0 * 128 + i];
    __syncthreads();

    const int j = tid;
    float s[32];
#pragma unroll
    for (int bb = 0; bb < 32; ++bb) s[bb] = B1[j];
    for (int i = 0; i < 128; ++i) {
        float w = W1[(size_t)i * 128 + j];
#pragma unroll
        for (int bb = 0; bb < 32; ++bb) s[bb] += xe[bb][i] * w;
    }
#pragma unroll
    for (int bb = 0; bb < 32; ++bb) {
        float v = s[bb];
        h1[bb][j] = v / (1.0f + expf(-v));   // silu
    }
    __syncthreads();

#pragma unroll
    for (int bb = 0; bb < 32; ++bb) s[bb] = B2[j];
    for (int i = 0; i < 128; ++i) {
        float w = W2[(size_t)i * 128 + j];
#pragma unroll
        for (int bb = 0; bb < 32; ++bb) s[bb] += h1[bb][i] * w;
    }
#pragma unroll
    for (int bb = 0; bb < 32; ++bb) {
        float v = s[bb];
        H[(size_t)(b0 + bb) * 128 + j] = v / (1.0f + expf(-v));
    }
}

// ---------------------------------------------------------------------------
// Fused W3-GEMM + einsum contraction.
// grid = (E/16, 12+5), block = 32 (one wave).
//   y <  12 : main, m = y      -> 4 d-blocks, 258 chunks, plain stores
//   y >= 12 : extra, m0 = y-12 -> 4 d-blocks, 258 chunks, atomicAdd
// A fragment (16x32 bf16): lane l -> row b = l%16;
//   slot s -> K(s) = s + 16*(l/16);  c = ch*32 + K(s).
// C/D (16x16 f32): lane l -> N = l%16 (=d), vgpr v -> M = v + 8*(l/16).
// ---------------------------------------------------------------------------
__global__ __launch_bounds__(32)
void so2_gemm(const float* __restrict__ X, const float* __restrict__ H,
              const __bf16* __restrict__ MP, const __bf16* __restrict__ EP,
              float* __restrict__ OutMain, float* __restrict__ OutExtra)
{
    const int lane  = threadIdx.x;
    const int btile = blockIdx.x;
    const int yy    = blockIdx.y;
    const int row   = lane & 15;
    const int half  = lane >> 4;
    const int b     = btile * 16 + row;

    const bool isMain = (yy < M_);
    const int  xm     = isMain ? yy : (yy - M_);   // m (main) or m0 (extra)

    // Per-lane x slice, pre-permuted into A-fragment slot order.
    const float* xrow = X + ((size_t)b * M_ + xm) * C_;
    float xv[32];
#pragma unroll
    for (int s = 0; s < 16; ++s) {
        int K = s + 16 * half;
        xv[s]      = xrow[K];        // chunk-half 0 (c in [0,32))
        xv[16 + s] = xrow[32 + K];   // chunk-half 1 (c in [32,64))
    }

    v8f acc[4] = {};

    const float*  hrow     = H + (size_t)b * EC_;
    const __bf16* base     = isMain ? (MP + (size_t)yy * 4 * NCH_M * CHELEMS) : EP;
    const size_t  nchPerDb = isMain ? NCH_M : NCH_E;
    const int     chunk0   = isMain ? 0 : xm * NCH_M;
    const __bf16* lbase    = base + lane * 16;

    // fragment pointer: chunk-major, then [lane][slot]
#define FRAGP(db_, chunk_) \
    ((const v16bf*)(lbase + ((size_t)(db_) * nchPerDb + (size_t)(chunk_)) * CHELEMS))

    v16bf bA[4], bB[4];        // ping-pong B-fragment buffers
#pragma unroll
    for (int db = 0; db < 4; ++db) bA[db] = *FRAGP(db, chunk0);   // preload ch0,k0

    for (int k = 0; k <= EC_; ++k) {           // k==128: virtual bias row
        const float hf = (k < EC_) ? hrow[k] : 1.0f;

        // ---- chunk-half 0: compute with bA, prefetch half 1 into bB ----
        v16bf a0;
#pragma unroll
        for (int s = 0; s < 16; ++s) a0[s] = (__bf16)(hf * xv[s]);
#pragma unroll
        for (int db = 0; db < 4; ++db) bB[db] = *FRAGP(db, chunk0 + k * 2 + 1);
#pragma unroll
        for (int db = 0; db < 4; ++db)
            acc[db] = __builtin_amdgcn_wmma_f32_16x16x32_bf16(
                false, a0, false, bA[db], (short)0, acc[db], false, false);

        // ---- chunk-half 1: compute with bB, prefetch next k into bA ----
        v16bf a1;
#pragma unroll
        for (int s = 0; s < 16; ++s) a1[s] = (__bf16)(hf * xv[16 + s]);
        if (k < EC_) {
#pragma unroll
            for (int db = 0; db < 4; ++db) bA[db] = *FRAGP(db, chunk0 + (k + 1) * 2);
        }
#pragma unroll
        for (int db = 0; db < 4; ++db)
            acc[db] = __builtin_amdgcn_wmma_f32_16x16x32_bf16(
                false, a1, false, bB[db], (short)0, acc[db], false, false);
    }
#undef FRAGP

    if (isMain) {
#pragma unroll
        for (int db = 0; db < 4; ++db) {
            int d = db * 16 + row;
#pragma unroll
            for (int v = 0; v < 8; ++v) {
                int bo = btile * 16 + v + 8 * half;
                OutMain[((size_t)bo * M_ + yy) * D_ + d] = acc[db][v];
            }
        }
    } else {
#pragma unroll
        for (int db = 0; db < 4; ++db) {
            int d = db * 16 + row;
#pragma unroll
            for (int v = 0; v < 8; ++v) {
                int bo = btile * 16 + v + 8 * half;
                atomicAdd(&OutExtra[(size_t)bo * XD_ + d], acc[db][v]);
            }
        }
    }
}

// ---------------------------------------------------------------------------
extern "C" void kernel_launch(void* const* d_in, const int* in_sizes, int n_in,
                              void* d_out, int out_size, void* d_ws, size_t ws_size,
                              hipStream_t stream)
{
    (void)in_sizes; (void)n_in; (void)out_size; (void)ws_size;

    const float* x      = (const float*)d_in[0];
    const float* x_edge = (const float*)d_in[1];
    const float* W1     = (const float*)d_in[2];
    const float* b1     = (const float*)d_in[3];
    const float* W2     = (const float*)d_in[4];
    const float* b2     = (const float*)d_in[5];
    const float* W3     = (const float*)d_in[6];
    const float* b3     = (const float*)d_in[7];

    float*  H  = (float*)((char*)d_ws + OFF_H);
    __bf16* MP = (__bf16*)((char*)d_ws + OFF_MP);
    __bf16* EP = (__bf16*)((char*)d_ws + OFF_EP);

    float* outMain  = (float*)d_out;
    float* outExtra = outMain + (size_t)E_ * M_ * D_;

    // 1) repack W3 (+b3 row) into bf16 WMMA-fragment order (~19 MB, L2-resident)
    {
        size_t total  = MPELEMS + EPELEMS;
        int    blocks = (int)((total + 255) / 256);
        so2_pack_w3<<<blocks, 256, 0, stream>>>(W3, b3, MP, EP);
    }
    // 2) edge MLP -> h (fp32)
    so2_mlp<<<E_ / 32, 128, 0, stream>>>(x_edge, W1, b1, W2, b2, H);
    // 3) zero the atomically-accumulated gating output
    hipMemsetAsync(outExtra, 0, (size_t)E_ * XD_ * sizeof(float), stream);
    // 4) fused 73-GFLOP bf16-WMMA GEMM + contraction
    dim3 grid(E_ / 16, M_ + M0_);
    so2_gemm<<<grid, 32, 0, stream>>>(x, H, MP, EP, outMain, outExtra);
}
